// Decode_29875792511742
// MI455X (gfx1250) — matrix operand order repaired and checked by reading, not compiled, not run on hardware
//
#include <hip/hip_runtime.h>
#include <hip/hip_bf16.h>

// ---------------------------------------------------------------- constants
#define B_      8
#define C_IN_   64
#define H_      128
#define W_      128
#define NPT_    128           // NUM_POINT
#define NI_     512           // N_INST
#define DM_     256           // D_MODEL
#define INIT_STRIDE_   10.0f
#define COARSE_STRIDE_ 4.0f
#define DOWN_SAMPLE_   4.0f

typedef _Float16 half_t;
typedef __attribute__((ext_vector_type(16))) _Float16 v16h;
typedef __attribute__((ext_vector_type(8)))  float    v8f;
typedef __attribute__((ext_vector_type(4)))  unsigned u32x4;
typedef __attribute__((ext_vector_type(8)))  int      i32x8;
typedef __attribute__((ext_vector_type(4)))  int      i32x4;

// ---------------------------------------------------------------- WMMA frag helpers
// CDNA5 wave32 layouts (CDNA5 ISA §7.12.2):
//  A (16x32 f16): lane&15 -> row, K = 16*(lane>>4) + i (contiguous order chosen here, not the ISA table order)
//  B (32x16 f16): lane&15 -> col, K = 16*(lane>>4) + i  (same contiguous order as A; contiguous in memory when B stored [N][K])
//  C/D (16x16 f32): lane&15 -> col, vgpr i -> row i + 8*(lane>>4)

__device__ __forceinline__ v16h frag_a(const half_t* A, int lda) {
  const int lane = threadIdx.x & 31;
  const int r = lane & 15;
  const int h = (lane >> 4) << 4;            // 0 or 16
  const half_t* p = A + r * lda + h;
  v16h f;
#pragma unroll
  for (int i = 0; i < 16; ++i) f[i] = p[i];
  return f;
}

__device__ __forceinline__ v16h frag_a_f32(const float* A, int lda) {
  const int lane = threadIdx.x & 31;
  const int r = lane & 15;
  const int h = (lane >> 4) << 4;
  const float* p = A + r * lda + h;
  v16h f;
#pragma unroll
  for (int i = 0; i < 16; ++i) f[i] = (half_t)p[i];
  return f;
}

// B stored transposed: BT[n][k], contiguous K per lane
__device__ __forceinline__ v16h frag_b(const half_t* BT, int ldb) {
  const int lane = threadIdx.x & 31;
  const int c = lane & 15;
  const int kh = (lane >> 4) << 4;           // 0 or 16
  const half_t* p = BT + c * ldb + kh;
  v16h f;
#pragma unroll
  for (int i = 0; i < 16; ++i) f[i] = p[i];
  return f;
}

// B stored row-major [K][N] (e.g. V matrix in LDS)
__device__ __forceinline__ v16h frag_b_rm(const half_t* Bm, int ldb) {
  const int lane = threadIdx.x & 31;
  const int c = lane & 15;
  const int kh = (lane >> 4) << 4;
  const half_t* p = Bm + (size_t)kh * ldb + c;
  v16h f;
#pragma unroll
  for (int i = 0; i < 16; ++i) f[i] = p[(size_t)i * ldb];
  return f;
}

__device__ __forceinline__ v8f wmma_f16(v16h a, v16h b, v8f c) {
  return __builtin_amdgcn_wmma_f32_16x16x32_f16(false, a, false, b, (short)0, c,
                                                false, false);
}

// ---------------------------------------------------------------- TDM bulk copy
// One tensor_load_to_lds moves a contiguous [n8 x 8B] tile global->LDS.
// D# per CDNA5 ISA: group0 = {count=1, lds_addr,
// global_addr, type=2}; group1 = {data_size=8B, tensor_dim0=tile_dim0=
// stride0=n8, tensor_dim1=tile_dim1=1}; groups 2/3 zero (<=2D tensor).
__device__ __forceinline__ void tdm_load_to_lds(unsigned lds_off,
                                                const void* gsrc,
                                                unsigned n8) {
  const unsigned long long ga = (unsigned long long)gsrc;
  u32x4 g0;
  g0[0] = 1u;                                   // count=1
  g0[1] = lds_off;                              // lds_addr (bytes)
  g0[2] = (unsigned)(ga & 0xffffffffu);         // global_addr[31:0]
  g0[3] = (unsigned)((ga >> 32) & 0x01ffffffu)  // global_addr[56:32]
          | (2u << 30);                         // type = 2 ("image")
  i32x8 g1;
  g1[0] = (3 << 16);                            // data_size = 8 bytes
  g1[1] = (int)((n8 & 0xffffu) << 16);          // tensor_dim0[15:0] @ bits 63:48
  g1[2] = (int)(((n8 >> 16) & 0xffffu)          // tensor_dim0[31:16]
                | (1u << 16));                  // tensor_dim1 = 1
  g1[3] = (int)((n8 & 0xffffu) << 16);          // tile_dim0 = n8 (<= 65535)
  g1[4] = 1;                                    // tile_dim1 = 1
  g1[5] = (int)n8;                              // tensor_dim0_stride = n8
  g1[6] = 0;
  g1[7] = 0;
  i32x4 gz = {0, 0, 0, 0};
#if __clang_major__ >= 23
  i32x8 gz8 = {0, 0, 0, 0, 0, 0, 0, 0};
  __builtin_amdgcn_tensor_load_to_lds(g0, g1, gz, gz, gz8, 0);
#else
  __builtin_amdgcn_tensor_load_to_lds(g0, g1, gz, gz, 0);
#endif
}

// ---------------------------------------------------------------- weight repack
__global__ __launch_bounds__(256) void repack_w1(const float* __restrict__ w,
                                                 half_t* __restrict__ wt) {
  // OIHW [256][64][3][3] -> [o][ (ky*3+kx)*64 + ic ]  (576 = K)
  for (int i = blockIdx.x * blockDim.x + threadIdx.x; i < 256 * 576;
       i += gridDim.x * blockDim.x) {
    const int o = i / 576, k = i % 576;
    const int tap = k >> 6, ic = k & 63;
    wt[i] = (half_t)w[(o * 64 + ic) * 9 + tap];
  }
}

__global__ __launch_bounds__(256) void repack_w2(const float* __restrict__ w,
                                                 half_t* __restrict__ wt) {
  // [254][256][1][1] -> [256 pad][256]
  for (int i = blockIdx.x * blockDim.x + threadIdx.x; i < 256 * 256;
       i += gridDim.x * blockDim.x) {
    const int o = i >> 8, ic = i & 255;
    wt[i] = (half_t)(o < 254 ? w[o * 256 + ic] : 0.f);
  }
}

__global__ __launch_bounds__(256) void repack_T(const float* __restrict__ w,
                                                half_t* __restrict__ wt,
                                                int In, int Out) {
  // math layout w[In][Out] (x @ w) -> wt[Out][In]
  for (int i = blockIdx.x * blockDim.x + threadIdx.x; i < In * Out;
       i += gridDim.x * blockDim.x) {
    const int o = i / In, ic = i % In;
    wt[i] = (half_t)w[ic * Out + o];
  }
}

// ---------------------------------------------------------------- init polys
__global__ __launch_bounds__(NPT_) void init_points_kernel(
    const float* __restrict__ wh, const int* __restrict__ ct_ind,
    const int* __restrict__ ct_img, float* __restrict__ points1,   // [NI][129][2]
    float* __restrict__ init_polys,                                // [NI][128][2]
    float* __restrict__ out0)                                      // [NI][128][2]
{
  const int inst = blockIdx.x;
  const int j = threadIdx.x;
  const int ind = ct_ind[inst];
  const int img = ct_img[inst];
  const int cx = ind % W_, cy = ind / W_;
  const float fx = (float)cx, fy = (float)cy;
  const size_t plane = (size_t)H_ * W_;
  const size_t base = (size_t)img * (2 * NPT_) * plane + (size_t)cy * W_ + cx;
  const float ox = wh[base + (size_t)(2 * j) * plane];
  const float oy = wh[base + (size_t)(2 * j + 1) * plane];
  const float px = ox * INIT_STRIDE_ + fx;
  const float py = oy * INIT_STRIDE_ + fy;
  const int ip = inst * NPT_ + j;
  init_polys[ip * 2 + 0] = px;
  init_polys[ip * 2 + 1] = py;
  out0[ip * 2 + 0] = px * DOWN_SAMPLE_;
  out0[ip * 2 + 1] = py * DOWN_SAMPLE_;
  float* prow = points1 + ((size_t)inst * 129 + 1 + j) * 2;
  prow[0] = px;
  prow[1] = py;
  if (j == 0) {
    points1[(size_t)inst * 129 * 2 + 0] = fx;
    points1[(size_t)inst * 129 * 2 + 1] = fy;
  }
}

// ---------------------------------------------------------------- conv 3x3 + ReLU
__global__ __launch_bounds__(256) void conv3x3_relu_kernel(
    const float* __restrict__ src,   // [B][64][H][W]
    const half_t* __restrict__ w1t,  // [256][576]
    const float* __restrict__ b1,    // [256]
    half_t* __restrict__ hidden)     // [B][H][W][256] f16
{
  __shared__ half_t patch[3][18][64];
  const int tilesPerRow = W_ / 16;
  const int tid = blockIdx.x;
  const int x0 = (tid % tilesPerRow) * 16;
  const int y = (tid / tilesPerRow) % H_;
  const int b = tid / (tilesPerRow * H_);

  __builtin_prefetch(w1t, 0, 1);
  for (int idx = threadIdx.x; idx < 3 * 18 * 64; idx += blockDim.x) {
    const int ic = idx & 63;
    const int col = idx >> 6;
    const int xx = col % 18;
    const int ky = col / 18;
    const int yy = y + ky - 1;
    const int xg = x0 + xx - 1;
    float v = 0.f;
    if (yy >= 0 && yy < H_ && xg >= 0 && xg < W_)
      v = src[(((size_t)b * C_IN_ + ic) * H_ + yy) * W_ + xg];
    patch[ky][xx][ic] = (half_t)v;
  }
  __syncthreads();

  const int wave = threadIdx.x >> 5;
  const int lane = threadIdx.x & 31;
  const int n = lane & 15;
  const int mh = (lane >> 4) << 3;
  const int nb = wave * 32;  // this wave's output-channel window [nb, nb+32)

  // one A fragment feeds two B fragments / two accumulators
  v8f acc0 = {}, acc1 = {};
#pragma unroll 2
  for (int ks = 0; ks < 18; ++ks) {
    const int kbase = ks * 32;
    const int tap = kbase >> 6;
    const int icb = kbase & 63;
    const int ky = tap / 3, kx = tap % 3;
    v16h a = frag_a(&patch[ky][kx][icb], 64);
    v16h bf0 = frag_b(w1t + nb * 576 + kbase, 576);
    v16h bf1 = frag_b(w1t + (nb + 16) * 576 + kbase, 576);
    acc0 = wmma_f16(a, bf0, acc0);
    acc1 = wmma_f16(a, bf1, acc1);
  }
  const float bias0 = b1[nb + n];
  const float bias1 = b1[nb + 16 + n];
  half_t* outBase = hidden + ((((size_t)b * H_ + y) * W_) + x0) * 256 + nb + n;
#pragma unroll
  for (int i = 0; i < 8; ++i) {
    float v0 = acc0[i] + bias0;
    float v1 = acc1[i] + bias1;
    v0 = v0 > 0.f ? v0 : 0.f;
    v1 = v1 > 0.f ? v1 : 0.f;
    outBase[(size_t)(mh + i) * 256] = (half_t)v0;
    outBase[(size_t)(mh + i) * 256 + 16] = (half_t)v1;
  }
}

// ---------------------------------------------------------------- conv 1x1 (+bias)
__global__ __launch_bounds__(256) void conv1x1_kernel(
    const half_t* __restrict__ hidden,  // [BHW][256]
    const half_t* __restrict__ w2t,     // [256][256]
    const float* __restrict__ b2,       // [254]
    half_t* __restrict__ feat)          // [BHW][256] (ch 254,255 zero)
{
  const size_t pix0 = (size_t)blockIdx.x * 16;
  const int wave = threadIdx.x >> 5;
  const int lane = threadIdx.x & 31;
  const int n = lane & 15;
  const int mh = (lane >> 4) << 3;
  const int nb = wave * 32;
  const half_t* A0 = hidden + pix0 * 256;
  __builtin_prefetch(A0, 0, 1);

  v8f acc0 = {}, acc1 = {};
#pragma unroll 2
  for (int ks = 0; ks < 8; ++ks) {
    v16h a = frag_a(A0 + ks * 32, 256);
    v16h bf0 = frag_b(w2t + nb * 256 + ks * 32, 256);
    v16h bf1 = frag_b(w2t + (nb + 16) * 256 + ks * 32, 256);
    acc0 = wmma_f16(a, bf0, acc0);
    acc1 = wmma_f16(a, bf1, acc1);
  }
#pragma unroll
  for (int t = 0; t < 2; ++t) {
    const int oc = nb + t * 16 + n;
    const float bias = (oc < 254) ? b2[oc] : 0.f;
    const v8f& acc = t ? acc1 : acc0;
    half_t* outB = feat + pix0 * 256 + oc;
#pragma unroll
    for (int i = 0; i < 8; ++i) {
      float v = (oc < 254) ? (acc[i] + bias) : 0.f;
      outB[(size_t)(mh + i) * 256] = (half_t)v;
    }
  }
}

// ---------------------------------------------------------------- sample + build x
__device__ __forceinline__ float samp(const half_t* feat, int img, int yy,
                                      int xx, float wgt, int c) {
  if (xx < 0 || xx >= W_ || yy < 0 || yy >= H_) return 0.f;
  return wgt * (float)feat[((((size_t)img * H_ + yy) * W_) + xx) * 256 + c];
}

__global__ __launch_bounds__(256) void sample_kernel(
    const half_t* __restrict__ feat, const int* __restrict__ ct_img,
    const float* __restrict__ pts,  // [NI][P][2]
    half_t* __restrict__ xbuf,      // [NI][Ppad][256]
    int P, int Ppad)
{
  __shared__ float sx[160], sy[160], nx[160], ny[160];
  __shared__ float s_mx, s_my, s_scale;
  const int inst = blockIdx.x;
  const int img = ct_img[inst];
  for (int p = threadIdx.x; p < Ppad; p += blockDim.x) {
    float px = 0.f, py = 0.f;
    if (p < P) {
      px = pts[((size_t)inst * P + p) * 2 + 0];
      py = pts[((size_t)inst * P + p) * 2 + 1];
    }
    sx[p] = px; sy[p] = py;
  }
  __syncthreads();
  if (threadIdx.x == 0) {
    float mx = 0.f, my = 0.f;
    for (int p = 0; p < P; ++p) { mx += sx[p]; my += sy[p]; }
    mx /= (float)P; my /= (float)P;
    float sc = 0.f;
    for (int p = 0; p < P; ++p) {
      sc = fmaxf(sc, fabsf(sx[p] - mx));
      sc = fmaxf(sc, fabsf(sy[p] - my));
    }
    s_mx = mx; s_my = my; s_scale = sc + 1e-6f;
  }
  __syncthreads();
  for (int p = threadIdx.x; p < Ppad; p += blockDim.x) {
    nx[p] = (sx[p] - s_mx) / s_scale;
    ny[p] = (sy[p] - s_my) / s_scale;
  }
  __syncthreads();

  for (int p = 0; p < Ppad; ++p) {
    half_t* dst = xbuf + ((size_t)inst * Ppad + p) * 256;
    if (p >= P) {
      for (int c = threadIdx.x; c < 256; c += blockDim.x) dst[c] = (half_t)0.f;
      continue;
    }
    const float ix = sx[p] - 0.5f, iy = sy[p] - 0.5f;
    const float x0f = floorf(ix), y0f = floorf(iy);
    const float wx = ix - x0f, wy = iy - y0f;
    const int xi = (int)x0f, yi = (int)y0f;
    const float w00 = (1.f - wx) * (1.f - wy), w10 = wx * (1.f - wy);
    const float w01 = (1.f - wx) * wy, w11 = wx * wy;
    for (int c = threadIdx.x; c < 256; c += blockDim.x) {
      float v;
      if (c < 254) {
        v = samp(feat, img, yi, xi, w00, c) + samp(feat, img, yi, xi + 1, w10, c) +
            samp(feat, img, yi + 1, xi, w01, c) +
            samp(feat, img, yi + 1, xi + 1, w11, c);
      } else {
        v = (c == 254) ? nx[p] : ny[p];
      }
      dst[c] = (half_t)v;
    }
  }
}

// ---------------------------------------------------------------- fused CSA
__global__ __launch_bounds__(256) void csa_kernel(
    const half_t* __restrict__ xg,      // [NI][Ppad][256]
    const half_t* __restrict__ wqkvt,   // [768][256]
    const float* __restrict__ bqkv,     // [768]
    const half_t* __restrict__ wot,     // [256][256]
    const float* __restrict__ bo,       // [256]
    const float* __restrict__ whead,    // [256][2]
    const float* __restrict__ bhead,    // [2]
    const float* __restrict__ poly_base,// [NI][128][2]
    float* __restrict__ out_sec,        // [NI][128][2]
    float* __restrict__ polys_next,     // [NI][128][2] or null
    int P, int Ppad, int Kpad, int skip_first, float stride_mul)
{
  extern __shared__ __align__(16) char smem[];
  half_t* q = (half_t*)smem;                      // [Ppad][256]  (reused as o)
  half_t* kmat = q + (size_t)Ppad * 256;          // [Ppad][256]  (reused as out)
  half_t* v = kmat + (size_t)Ppad * 256;          // [Kpad][256]
  char* scratch = (char*)(v + (size_t)Kpad * 256);
  half_t* x = (half_t*)scratch;                   // [Ppad][256] (phase 1-2)
  float* S = (float*)scratch;                     // [Ppad][Kpad] (phase 3)

  const int inst = blockIdx.x;
  const int tid = threadIdx.x;
  const int wave = tid >> 5;
  const int lane = tid & 31;
  const int n = lane & 15;
  const int mh = (lane >> 4) << 3;
  const int mt = Ppad / 16;

  // ---- TDM: one tensor DMA pulls the whole x tile (Ppad*512 B) into LDS
  const half_t* xsrc = xg + (size_t)inst * Ppad * 256;
  if (wave == 0) {
    tdm_load_to_lds((unsigned)(size_t)(void*)x, xsrc,
                    (unsigned)(Ppad * 256 * 2 / 8));
    __builtin_amdgcn_s_wait_tensorcnt(0);
  }
  // zero V padding rows while the DMA target region is independent
  for (int i = tid; i < (Kpad - Ppad) * 256; i += 256)
    v[(size_t)Ppad * 256 + i] = (half_t)0.f;
  __syncthreads();

  // ---- QKV: [Ppad][256] x [256][768]
  const int njobs = 3 * 16 * mt;
  for (int job = wave; job < njobs; job += 8) {
    const int m = job % mt;
    const int ncol = (job / mt) % 16;
    const int sec = job / (mt * 16);
    const int ob = sec * 256 + ncol * 16;
    v8f acc = {};
#pragma unroll 1
    for (int ks = 0; ks < 8; ++ks) {
      v16h a = frag_a(x + (size_t)m * 16 * 256 + ks * 32, 256);
      v16h bf = frag_b(wqkvt + (size_t)ob * 256 + ks * 32, 256);
      acc = wmma_f16(a, bf, acc);
    }
    half_t* dstbuf = (sec == 0) ? q : (sec == 1) ? kmat : v;
    const float bias = bqkv[ob + n];
#pragma unroll
    for (int i = 0; i < 8; ++i)
      dstbuf[(size_t)(m * 16 + mh + i) * 256 + ncol * 16 + n] =
          (half_t)(acc[i] + bias);
  }
  __syncthreads();

  // ---- per-head attention (d = 32, 8 heads)
  const float scale = 0.17677669529663687f;  // 1/sqrt(32)
  for (int h = 0; h < 8; ++h) {
    // scores S = q_h @ k_h^T : one 16x16x32 WMMA per tile (K = 32 = head dim)
    for (int job = wave; job < mt * mt; job += 8) {
      const int m = job % mt, nc = job / mt;
      v16h a = frag_a(q + (size_t)m * 16 * 256 + h * 32, 256);
      v16h bf = frag_b(kmat + (size_t)nc * 16 * 256 + h * 32, 256); // k rows ARE B^T
      v8f acc = {};
      acc = wmma_f16(a, bf, acc);
      const int col = nc * 16 + n;
#pragma unroll
      for (int i = 0; i < 8; ++i) {
        float s = acc[i] * scale;
        if (col >= P) s = -1e30f;
        S[(size_t)(m * 16 + mh + i) * Kpad + col] = s;
      }
    }
    __syncthreads();
    // softmax rows (mask tail, zero pad cols up to Kpad)
    for (int r = tid; r < Ppad; r += 256) {
      float* row = S + (size_t)r * Kpad;
      float mx = -1e30f;
      for (int c = 0; c < P; ++c) mx = fmaxf(mx, row[c]);
      float sum = 0.f;
      for (int c = 0; c < P; ++c) {
        float e = __expf(row[c] - mx);
        row[c] = e;
        sum += e;
      }
      const float inv = 1.f / sum;
      for (int c = 0; c < P; ++c) row[c] *= inv;
      for (int c = P; c < Kpad; ++c) row[c] = 0.f;
    }
    __syncthreads();
    // o_h = S @ v_h, written over q columns [h*32, h*32+32)
    for (int job = wave; job < mt * 2; job += 8) {
      const int m = job % mt, nt = job / mt;
      v8f acc = {};
#pragma unroll 1
      for (int ks = 0; ks < Kpad / 32; ++ks) {
        v16h a = frag_a_f32(S + (size_t)m * 16 * Kpad + ks * 32, Kpad);
        v16h bf = frag_b_rm(v + (size_t)ks * 32 * 256 + h * 32 + nt * 16, 256);
        acc = wmma_f16(a, bf, acc);
      }
#pragma unroll
      for (int i = 0; i < 8; ++i)
        q[(size_t)(m * 16 + mh + i) * 256 + h * 32 + nt * 16 + n] =
            (half_t)acc[i];
    }
    __syncthreads();
  }

  // ---- out = o @ Wo + bo  (o lives in q buffer; result into kmat buffer)
  for (int job = wave; job < mt * 16; job += 8) {
    const int m = job % mt, nc = job / mt;
    v8f acc = {};
#pragma unroll 1
    for (int ks = 0; ks < 8; ++ks) {
      v16h a = frag_a(q + (size_t)m * 16 * 256 + ks * 32, 256);
      v16h bf = frag_b(wot + (size_t)(nc * 16) * 256 + ks * 32, 256);
      acc = wmma_f16(a, bf, acc);
    }
    const float bias = bo[nc * 16 + n];
#pragma unroll
    for (int i = 0; i < 8; ++i)
      kmat[(size_t)(m * 16 + mh + i) * 256 + nc * 16 + n] =
          (half_t)(acc[i] + bias);
  }
  __syncthreads();

  // ---- head projection (256 -> 2) + polygon update + scaled output
  const int Pout = P - skip_first;  // always 128 here
  for (int p = tid; p < Pout; p += 256) {
    const int row = p + skip_first;
    float ax = bhead[0], ay = bhead[1];
    const half_t* orow = kmat + (size_t)row * 256;
    for (int c = 0; c < 256; ++c) {
      const float o = (float)orow[c];
      ax += o * whead[c * 2 + 0];
      ay += o * whead[c * 2 + 1];
    }
    const size_t ip = (size_t)inst * Pout + p;
    const float bx = poly_base[ip * 2 + 0];
    const float by = poly_base[ip * 2 + 1];
    const float rx = ax * stride_mul + bx;
    const float ry = ay * stride_mul + by;
    out_sec[ip * 2 + 0] = rx * DOWN_SAMPLE_;
    out_sec[ip * 2 + 1] = ry * DOWN_SAMPLE_;
    if (polys_next) {
      polys_next[ip * 2 + 0] = rx;
      polys_next[ip * 2 + 1] = ry;
    }
  }
}

// ---------------------------------------------------------------- host launcher
static inline size_t align256(size_t x) { return (x + 255) & ~(size_t)255; }

extern "C" void kernel_launch(void* const* d_in, const int* in_sizes, int n_in,
                              void* d_out, int out_size, void* d_ws,
                              size_t ws_size, hipStream_t stream) {
  const float* cnn = (const float*)d_in[0];
  const float* wh = (const float*)d_in[1];
  const int* ct_ind = (const int*)d_in[2];
  const int* ct_img = (const int*)d_in[3];
  const float* f1_w1 = (const float*)d_in[4];
  const float* f1_b1 = (const float*)d_in[5];
  const float* f1_w2 = (const float*)d_in[6];
  const float* f1_b2 = (const float*)d_in[7];
  const float* f2_w1 = (const float*)d_in[8];
  const float* f2_b1 = (const float*)d_in[9];
  const float* f2_w2 = (const float*)d_in[10];
  const float* f2_b2 = (const float*)d_in[11];
  const float* c1_wqkv = (const float*)d_in[12];
  const float* c1_bqkv = (const float*)d_in[13];
  const float* c1_wo = (const float*)d_in[14];
  const float* c1_bo = (const float*)d_in[15];
  const float* c1_wh = (const float*)d_in[16];
  const float* c1_bh = (const float*)d_in[17];
  const float* c2_wqkv = (const float*)d_in[18];
  const float* c2_bqkv = (const float*)d_in[19];
  const float* c2_wo = (const float*)d_in[20];
  const float* c2_bo = (const float*)d_in[21];
  const float* c2_wh = (const float*)d_in[22];
  const float* c2_bh = (const float*)d_in[23];

  float* out = (float*)d_out;
  float* out0 = out;                       // init_polys * 4
  float* out1 = out + NI_ * NPT_ * 2;      // coarse * 4
  float* out2 = out + 2 * NI_ * NPT_ * 2;  // final * 4

  // ---- workspace carve
  char* ws = (char*)d_ws;
  size_t off = 0;
  auto carve = [&](size_t bytes) {
    char* p = ws + off;
    off += align256(bytes);
    return p;
  };
  const size_t NPIX = (size_t)B_ * H_ * W_;
  half_t* w1t_a = (half_t*)carve(256 * 576 * 2);
  half_t* w1t_b = (half_t*)carve(256 * 576 * 2);
  half_t* w2t_a = (half_t*)carve(256 * 256 * 2);
  half_t* w2t_b = (half_t*)carve(256 * 256 * 2);
  half_t* wqkvt1 = (half_t*)carve(768 * 256 * 2);
  half_t* wqkvt2 = (half_t*)carve(768 * 256 * 2);
  half_t* wot1 = (half_t*)carve(256 * 256 * 2);
  half_t* wot2 = (half_t*)carve(256 * 256 * 2);
  half_t* hidden = (half_t*)carve(NPIX * 256 * 2);
  half_t* feat1 = (half_t*)carve(NPIX * 256 * 2);
  half_t* feat2 = (half_t*)carve(NPIX * 256 * 2);
  float* points1 = (float*)carve((size_t)NI_ * 129 * 2 * 4);
  float* init_polys = (float*)carve((size_t)NI_ * 128 * 2 * 4);
  float* coarse = (float*)carve((size_t)NI_ * 128 * 2 * 4);
  half_t* xbuf = (half_t*)carve((size_t)NI_ * 144 * 256 * 2);
  (void)ws_size; (void)n_in; (void)in_sizes; (void)out_size;

  // ---- 1. weight repacks
  repack_w1<<<144, 256, 0, stream>>>(f1_w1, w1t_a);
  repack_w1<<<144, 256, 0, stream>>>(f2_w1, w1t_b);
  repack_w2<<<64, 256, 0, stream>>>(f1_w2, w2t_a);
  repack_w2<<<64, 256, 0, stream>>>(f2_w2, w2t_b);
  repack_T<<<192, 256, 0, stream>>>(c1_wqkv, wqkvt1, 256, 768);
  repack_T<<<192, 256, 0, stream>>>(c2_wqkv, wqkvt2, 256, 768);
  repack_T<<<64, 256, 0, stream>>>(c1_wo, wot1, 256, 256);
  repack_T<<<64, 256, 0, stream>>>(c2_wo, wot2, 256, 256);

  // ---- 2. centers + init polys (+ output 0)
  init_points_kernel<<<NI_, NPT_, 0, stream>>>(wh, ct_ind, ct_img, points1,
                                               init_polys, out0);

  // ---- 3-6. trans_feature for both branches (hidden buffer reused)
  const int convBlocks = (int)(NPIX / 16);
  conv3x3_relu_kernel<<<convBlocks, 256, 0, stream>>>(cnn, w1t_a, f1_b1, hidden);
  conv1x1_kernel<<<convBlocks, 256, 0, stream>>>(hidden, w2t_a, f1_b2, feat1);
  conv3x3_relu_kernel<<<convBlocks, 256, 0, stream>>>(cnn, w1t_b, f2_b1, hidden);
  conv1x1_kernel<<<convBlocks, 256, 0, stream>>>(hidden, w2t_b, f2_b2, feat2);

  // ---- 7-8. stage 1: sample feat1 at 129 points, fused CSA
  {
    const int P = 129, Ppad = 144, Kpad = 160;
    sample_kernel<<<NI_, 256, 0, stream>>>(feat1, ct_img, points1, xbuf, P, Ppad);
    const size_t qb = (size_t)Ppad * 256 * 2;
    const size_t vb = (size_t)Kpad * 256 * 2;
    const size_t sb = (size_t)Ppad * Kpad * 4;
    const size_t shmem = qb * 2 + vb + (sb > qb ? sb : qb);
    csa_kernel<<<NI_, 256, shmem, stream>>>(
        xbuf, wqkvt1, c1_bqkv, wot1, c1_bo, c1_wh, c1_bh, init_polys, out1,
        coarse, P, Ppad, Kpad, /*skip_first=*/1, COARSE_STRIDE_);
  }

  // ---- 9-10. stage 2: sample feat2 at coarse polys, fused CSA
  {
    const int P = 128, Ppad = 128, Kpad = 128;
    sample_kernel<<<NI_, 256, 0, stream>>>(feat2, ct_img, coarse, xbuf, P, Ppad);
    const size_t qb = (size_t)Ppad * 256 * 2;
    const size_t vb = (size_t)Kpad * 256 * 2;
    const size_t sb = (size_t)Ppad * Kpad * 4;
    const size_t shmem = qb * 2 + vb + (sb > qb ? sb : qb);
    csa_kernel<<<NI_, 256, shmem, stream>>>(
        xbuf, wqkvt2, c2_bqkv, wot2, c2_bo, c2_wh, c2_bh, coarse, out2,
        (float*)nullptr, P, Ppad, Kpad, /*skip_first=*/0, 1.0f);
  }
}
